// MAGClassifier_9345848836359
// MI455X (gfx1250) — matrix operand order repaired and checked by reading, not compiled, not run on hardware
//
#include <hip/hip_runtime.h>
#include <hip/hip_bf16.h>

typedef __attribute__((ext_vector_type(16))) _Float16 v16h;
typedef __attribute__((ext_vector_type(8)))  _Float16 v8h;
typedef __attribute__((ext_vector_type(8)))  float    v8f;

#define NN    1024   // nodes
#define LE    4096   // edges
#define DIMC  256
#define HIDC  128
#define NHEAD 8
#define DHC   32
#define NGR   16
#define SCALE_QK 0.17677669529663687f   // 1/sqrt(32)

// 16-bit A-fragment K index (ISA 7.12.2): lo 8 elems = k0+half*8.., hi 8 = +16
__device__ __forceinline__ int frag_ka(int e, int half) {
  return half * 8 + (e & 7) + ((e & 8) << 1);
}

__device__ __forceinline__ v16h combine16(v8h lo, v8h hi) {
  v16h r;
#pragma unroll
  for (int i = 0; i < 8; ++i) { r[i] = lo[i]; r[i + 8] = hi[i]; }
  return r;
}
// A-fragment: base points at row start + k0; 16 contiguous K values at half*16
__device__ __forceinline__ v16h load_afrag(const _Float16* base, int half) {
  const _Float16* p = base + half * 16;
  return combine16(*(const v8h*)p, *(const v8h*)(p + 8));
}
// B-fragment: 16 contiguous K values at half*16 (lanes 0-15: K0-15, 16-31: K16-31)
__device__ __forceinline__ v16h load_bfrag(const _Float16* base, int half) {
  const _Float16* p = base + half * 16;
  return combine16(*(const v8h*)p, *(const v8h*)(p + 8));
}

// ---------------------------------------------------------------------------
// Weight prep: W32[K=256][N=256] row-major -> Wt16[N][K] (f16, column-major)
// ---------------------------------------------------------------------------
__global__ __launch_bounds__(256) void wprep_k(const float* __restrict__ W,
                                               _Float16* __restrict__ Wt)
{
  int id = blockIdx.x * 256 + threadIdx.x;
  int n = id >> 8, k = id & 255;
  Wt[n * DIMC + k] = (_Float16)W[k * DIMC + n];
}

// ---------------------------------------------------------------------------
// WMMA GEMM: C[M,N] = A16[M,K] @ Wt16[N,K]^T + bias. One wave per 16x64 tile
// (one A fragment shared by 4 B tiles -> 4 consecutive WMMAs per K step).
// Output targets selected at compile time (no epilogue branches).
// ---------------------------------------------------------------------------
template <bool OUT16, bool OUT32, bool RELU, bool OUT16T>
__global__ __launch_bounds__(32) void gemm_tile64(
    const _Float16* __restrict__ A, const _Float16* __restrict__ Wt,
    const float* __restrict__ bias,
    _Float16* __restrict__ out16, float* __restrict__ out32,
    _Float16* __restrict__ out16t, int M, int N, int K)
{
  int tilesN = N >> 6;
  int tm = (blockIdx.x / tilesN) << 4;
  int tn = (blockIdx.x % tilesN) << 6;
  int lane = threadIdx.x & 31;
  int half = lane >> 4;
  int mn = lane & 15;

  v8f acc[4] = {v8f{}, v8f{}, v8f{}, v8f{}};
  const _Float16* arow = &A[(tm + mn) * K];
  for (int k0 = 0; k0 < K; k0 += 32) {
    v16h a = load_afrag(arow + k0, half);
#pragma unroll
    for (int j = 0; j < 4; ++j) {
      v16h b = load_bfrag(&Wt[(tn + j * 16 + mn) * K + k0], half);
      acc[j] = __builtin_amdgcn_wmma_f32_16x16x32_f16(false, a, false, b,
                                                      (short)0, acc[j], false, false);
    }
  }
#pragma unroll
  for (int j = 0; j < 4; ++j) {
    int n = tn + j * 16 + mn;
    float bv = bias[n];
#pragma unroll
    for (int r = 0; r < 8; ++r) {
      int m = tm + r + (half << 3);
      float v = acc[j][r] + bv;
      if (RELU) v = fmaxf(v, 0.0f);
      if (OUT32)  out32[m * N + n] = v;
      if (OUT16)  out16[m * N + n] = (_Float16)v;
      if (OUT16T) out16t[n * M + m] = (_Float16)v;
    }
  }
}

// ---------------------------------------------------------------------------
// Flash attention with adjacency mask. One wave per (head, 16-row q tile).
// S^T = K * Q^T per 32-key chunk (2 WMMAs), online softmax per q column
// (= lane), then O^T += V^T * P^T (2 WMMAs). V supplied transposed
// (vt[d][key]) so all fragments are contiguous b128 loads.
// ---------------------------------------------------------------------------
__global__ __launch_bounds__(32) void flash_attn_k(
    const _Float16* __restrict__ q16, const _Float16* __restrict__ k16,
    const _Float16* __restrict__ vt, const int* __restrict__ srcA,
    const int* __restrict__ dstA, float* __restrict__ outO)
{
  __shared__ int s_src[LE];
  __shared__ int s_dst[LE];
  const int lane = threadIdx.x;
  for (int i = lane; i < LE; i += 32) { s_src[i] = srcA[i]; s_dst[i] = dstA[i]; }
  __syncthreads();

  const int head = blockIdx.y;
  const int qt   = blockIdx.x;
  const int hoff = head * DHC;
  const int half = lane >> 4;
  const int n    = lane & 15;      // q column of S^T / O^T
  const int qi   = qt * 16 + n;
  const int qs = s_src[qi], qd = s_dst[qi];

  // Q^T as B-fragment: 16 contiguous dh values
  v16h bq = load_bfrag(&q16[qi * DIMC + hoff], half);

  float m_run = -3.4e38f, l_run = 0.f;
  v8f o0 = {}, o1 = {};

  for (int c = 0; c < LE; c += 32) {
    if (c + 32 < LE) {   // prefetch next chunk (global_prefetch_b8)
      __builtin_prefetch(&k16[(c + 32 + lane) * DIMC + hoff], 0, 1);
      __builtin_prefetch(&vt[(hoff + lane) * LE + c + 32], 0, 1);
    }
    // K tiles as A-fragments: A[key][dh], contiguous dh
    v16h ka0 = load_afrag(&k16[(c + n) * DIMC + hoff], half);
    v16h ka1 = load_afrag(&k16[(c + 16 + n) * DIMC + hoff], half);
    v8f s0 = {}, s1 = {};
    s0 = __builtin_amdgcn_wmma_f32_16x16x32_f16(false, ka0, false, bq, (short)0, s0, false, false);
    s1 = __builtin_amdgcn_wmma_f32_16x16x32_f16(false, ka1, false, bq, (short)0, s1, false, false);

    // lane holds s0[r]=score(key=c+r+8*half, q=qi), s1[r]=key+16
    float p0[8], p1[8];
    float tmax = -3.4e38f;
#pragma unroll
    for (int r = 0; r < 8; ++r) {
      int k0i = c + r + 8 * half;
      int k1i = k0i + 16;
      bool m0 = (qs == s_src[k0i]) | (qs == s_dst[k0i]) | (qd == s_src[k0i]) | (qd == s_dst[k0i]);
      bool m1 = (qs == s_src[k1i]) | (qs == s_dst[k1i]) | (qd == s_src[k1i]) | (qd == s_dst[k1i]);
      p0[r] = m0 ? s0[r] * SCALE_QK : -1.0e9f;
      p1[r] = m1 ? s1[r] * SCALE_QK : -1.0e9f;
      tmax = fmaxf(tmax, fmaxf(p0[r], p1[r]));
    }
    tmax = fmaxf(tmax, __shfl_xor(tmax, 16, 32));   // combine halves (same q col)
    float m_new = fmaxf(m_run, tmax);
    float alpha = __expf(m_run - m_new);
    l_run *= alpha;
#pragma unroll
    for (int r = 0; r < 8; ++r) { o0[r] *= alpha; o1[r] *= alpha; }

    float tsum = 0.f;
#pragma unroll
    for (int r = 0; r < 8; ++r) {
      p0[r] = __expf(p0[r] - m_new);
      p1[r] = __expf(p1[r] - m_new);
      tsum += p0[r] + p1[r];
    }
    tsum += __shfl_xor(tsum, 16, 32);
    l_run += tsum;
    m_run = m_new;

    // Build P^T B-fragment: lane half h needs keys h*16+e for e=0..15.
    // Own half holds keys {h*8+r, 16+h*8+r}; partner (xor 16) supplies rest.
    v16h pb;
#pragma unroll
    for (int r = 0; r < 8; ++r) {
      float x0 = __shfl_xor(p0[r], 16, 32);
      float x1 = __shfl_xor(p1[r], 16, 32);
      if (half == 0) {
        pb[r]     = (_Float16)p0[r];  // keys 0-7
        pb[r + 8] = (_Float16)x0;     // keys 8-15 (partner acc0)
      } else {
        pb[r]     = (_Float16)x1;     // keys 16-23 (partner acc1)
        pb[r + 8] = (_Float16)p1[r];  // keys 24-31
      }
    }

    // V^T tiles as A-fragments: A[dh][key], contiguous keys (vt layout)
    v16h va0 = load_afrag(&vt[(hoff + n) * LE + c], half);
    v16h va1 = load_afrag(&vt[(hoff + 16 + n) * LE + c], half);
    o0 = __builtin_amdgcn_wmma_f32_16x16x32_f16(false, va0, false, pb, (short)0, o0, false, false);
    o1 = __builtin_amdgcn_wmma_f32_16x16x32_f16(false, va1, false, pb, (short)0, o1, false, false);
  }

  float inv = 1.0f / l_run;
#pragma unroll
  for (int r = 0; r < 8; ++r) {
    int q = qt * 16 + n;
    outO[q * DIMC + hoff + r + 8 * half]      = o0[r] * inv;
    outO[q * DIMC + hoff + 16 + r + 8 * half] = o1[r] * inv;
  }
}

// ---------------------------------------------------------------------------
// out = LayerNorm(A + Badd) * gamma + beta ; one wave per row (256 dims).
// ---------------------------------------------------------------------------
__global__ __launch_bounds__(32) void add_ln_k(
    const float* __restrict__ A, const float* __restrict__ Badd,
    const float* __restrict__ gamma, const float* __restrict__ beta,
    float* __restrict__ out32, _Float16* __restrict__ out16)
{
  int row = blockIdx.x, lane = threadIdx.x;
  float v[8];
  float s = 0.f;
#pragma unroll
  for (int i = 0; i < 8; ++i) {
    int d = lane + i * 32;
    v[i] = A[row * DIMC + d] + Badd[row * DIMC + d];
    s += v[i];
  }
#pragma unroll
  for (int m = 16; m >= 1; m >>= 1) s += __shfl_xor(s, m, 32);
  float mu = s * (1.0f / 256.0f);
  float var = 0.f;
#pragma unroll
  for (int i = 0; i < 8; ++i) { float t = v[i] - mu; var += t * t; }
#pragma unroll
  for (int m = 16; m >= 1; m >>= 1) var += __shfl_xor(var, m, 32);
  float rs = rsqrtf(var * (1.0f / 256.0f) + 1e-5f);
#pragma unroll
  for (int i = 0; i < 8; ++i) {
    int d = lane + i * 32;
    float hn = (v[i] - mu) * rs * gamma[d] + beta[d];
    out32[row * DIMC + d] = hn;
    out16[row * DIMC + d] = (_Float16)hn;
  }
}

// ---------------------------------------------------------------------------
// Encoders
// ---------------------------------------------------------------------------
__global__ __launch_bounds__(HIDC) void node_enc_k(
    const float* __restrict__ x, const float* __restrict__ W,
    const float* __restrict__ b, float* __restrict__ nodeF)
{
  int n = blockIdx.x, j = threadIdx.x;
  float a = b[j];
  for (int i = 0; i < 32; ++i) a += x[n * 32 + i] * W[i * HIDC + j];
  nodeF[n * HIDC + j] = a;
}

__global__ __launch_bounds__(DIMC) void edge_enc_k(
    const float* __restrict__ ea, const float* __restrict__ We,
    const float* __restrict__ be, const float* __restrict__ nodeF,
    const int* __restrict__ src, const int* __restrict__ dst,
    const int* __restrict__ batch, float* __restrict__ h32,
    _Float16* __restrict__ h16, int* __restrict__ ebatch)
{
  int e = blockIdx.x, c = threadIdx.x;
  int j = c & 127;
  float ef = be[j];
  for (int i = 0; i < 16; ++i) ef += ea[e * 16 + i] * We[i * HIDC + j];
  int node = (c < HIDC) ? src[e] : dst[e];
  float v = nodeF[node * HIDC + j] + ef;
  h32[e * DIMC + c] = v;
  h16[e * DIMC + c] = (_Float16)v;
  if (c == 0) ebatch[e] = batch[src[e]];
}

// ---------------------------------------------------------------------------
// PMA: q projection of the (shared) seed
// ---------------------------------------------------------------------------
__global__ __launch_bounds__(DIMC) void pma_q_k(
    const float* __restrict__ seed, const float* __restrict__ Wq,
    const float* __restrict__ bq, float* __restrict__ qp)
{
  int t = threadIdx.x;
  float a = bq[t];
  for (int i = 0; i < DIMC; ++i) a += seed[i] * Wq[i * DIMC + t];
  qp[t] = a;
}

// PMA attention: one wave per (graph, head); masked softmax over 4096 keys.
// V supplied transposed: vt[d][key].
__global__ __launch_bounds__(32) void pma_attn_k(
    const float* __restrict__ qp, const _Float16* __restrict__ k16,
    const _Float16* __restrict__ vt, const int* __restrict__ ebatch,
    float* __restrict__ Oa)
{
  __shared__ float sacc[32 * 32];
  __shared__ float sm[32], sl[32];
  int g = blockIdx.x, h = blockIdx.y, lane = threadIdx.x;
  int hoff = h * DHC;
  float qv[DHC];
#pragma unroll
  for (int d = 0; d < DHC; ++d) qv[d] = qp[hoff + d];
  float m_i = -3.4e38f, l_i = 0.f;
  float acc[DHC];
#pragma unroll
  for (int d = 0; d < DHC; ++d) acc[d] = 0.f;
  for (int k = lane; k < LE; k += 32) {
    if (ebatch[k] != g) continue;
    float sdot = 0.f;
#pragma unroll
    for (int d = 0; d < DHC; ++d) sdot += qv[d] * (float)k16[k * DIMC + hoff + d];
    sdot *= SCALE_QK;
    float mn = fmaxf(m_i, sdot);
    float al = __expf(m_i - mn);
    float p  = __expf(sdot - mn);
    l_i = l_i * al + p;
#pragma unroll
    for (int d = 0; d < DHC; ++d)
      acc[d] = acc[d] * al + p * (float)vt[(hoff + d) * LE + k];
    m_i = mn;
  }
  sm[lane] = m_i; sl[lane] = l_i;
#pragma unroll
  for (int d = 0; d < DHC; ++d) sacc[d * 32 + lane] = acc[d];
  __syncthreads();
  float m = -3.4e38f;
  for (int j = 0; j < 32; ++j) m = fmaxf(m, sm[j]);
  float lsum = 0.f, a = 0.f;
  for (int j = 0; j < 32; ++j) {
    float w = __expf(sm[j] - m);
    lsum += sl[j] * w;
    a += sacc[lane * 32 + j] * w;
  }
  Oa[g * DIMC + hoff + lane] = a / lsum;
}

// ---------------------------------------------------------------------------
// PMA tail: LN -> FFN -> LN -> cls1 (relu) -> cls2; one block per graph.
// ---------------------------------------------------------------------------
__device__ __forceinline__ float block_sum_256(float v, float* red) {
  int t = threadIdx.x;
  red[t] = v; __syncthreads();
  for (int s = 128; s > 0; s >>= 1) { if (t < s) red[t] += red[t + s]; __syncthreads(); }
  float r = red[0]; __syncthreads();
  return r;
}

__global__ __launch_bounds__(256) void pma_tail_k(
    const float* __restrict__ qp, const float* __restrict__ Oa,
    const float* __restrict__ g1, const float* __restrict__ b1,
    const float* __restrict__ g2, const float* __restrict__ b2,
    const float* __restrict__ Wo, const float* __restrict__ bo,
    const float* __restrict__ W1, const float* __restrict__ b1c,
    const float* __restrict__ W2, const float* __restrict__ b2c,
    float* __restrict__ out)
{
  __shared__ float red[256];
  __shared__ float H[256];
  __shared__ float G[256];
  int g = blockIdx.x, t = threadIdx.x;

  float hv = qp[t] + Oa[g * DIMC + t];
  float mu = block_sum_256(hv, red) * (1.0f / 256.0f);
  float dv = hv - mu;
  float var = block_sum_256(dv * dv, red) * (1.0f / 256.0f);
  float hn = dv * rsqrtf(var + 1e-5f) * g1[t] + b1[t];
  H[t] = hn; __syncthreads();

  float f = bo[t];
  for (int i = 0; i < DIMC; ++i) f += H[i] * Wo[i * DIMC + t];
  f = fmaxf(f, 0.0f);
  float h2 = hn + f;
  float mu2 = block_sum_256(h2, red) * (1.0f / 256.0f);
  float d2 = h2 - mu2;
  float var2 = block_sum_256(d2 * d2, red) * (1.0f / 256.0f);
  float gn = d2 * rsqrtf(var2 + 1e-5f) * g2[t] + b2[t];
  G[t] = gn; __syncthreads();

  float c1 = 0.f;
  if (t < HIDC) {
    c1 = b1c[t];
    for (int i = 0; i < DIMC; ++i) c1 += G[i] * W1[i * HIDC + t];
    c1 = fmaxf(c1, 0.0f) * W2[t];
  }
  float logit = block_sum_256(c1, red);
  if (t == 0) out[g] = logit + b2c[0];
}

// ---------------------------------------------------------------------------
// Host orchestration
// ---------------------------------------------------------------------------
struct MabIdx { int b1, b2, g1, g2, kW, kb, oW, ob, qW, qb, vW, vb; };

enum GemmKind { G_Q, G_K, G_VT, G_FFN };

static void launch_gemm(hipStream_t s, GemmKind kind,
                        const _Float16* A, const float* W, _Float16* Wt,
                        const float* bias, _Float16* o16, float* o32,
                        _Float16* o16t, int M, int N, int K) {
  wprep_k<<<(N * K) / 256, 256, 0, s>>>(W, Wt);
  dim3 grid((M / 16) * (N / 64));
  switch (kind) {
    case G_Q:
      gemm_tile64<true, true, false, false><<<grid, 32, 0, s>>>(A, Wt, bias, o16, o32, o16t, M, N, K);
      break;
    case G_K:
      gemm_tile64<true, false, false, false><<<grid, 32, 0, s>>>(A, Wt, bias, o16, o32, o16t, M, N, K);
      break;
    case G_VT:
      gemm_tile64<false, false, false, true><<<grid, 32, 0, s>>>(A, Wt, bias, o16, o32, o16t, M, N, K);
      break;
    case G_FFN:
      gemm_tile64<false, true, true, false><<<grid, 32, 0, s>>>(A, Wt, bias, o16, o32, o16t, M, N, K);
      break;
  }
}

static void run_sab(hipStream_t s, void* const* din, const MabIdx& mi,
                    float* h32, _Float16* h16, float* q32, _Float16* q16,
                    _Float16* k16v, _Float16* vt, _Float16* wtbuf, float* attnO,
                    const int* src, const int* dst)
{
  auto F = [&](int i) { return (const float*)din[i]; };
  launch_gemm(s, G_Q,  h16, F(mi.qW), wtbuf, F(mi.qb), q16, q32, nullptr, LE, DIMC, DIMC);
  launch_gemm(s, G_K,  h16, F(mi.kW), wtbuf, F(mi.kb), k16v, nullptr, nullptr, LE, DIMC, DIMC);
  launch_gemm(s, G_VT, h16, F(mi.vW), wtbuf, F(mi.vb), nullptr, nullptr, vt, LE, DIMC, DIMC);
  flash_attn_k<<<dim3(LE / 16, NHEAD), 32, 0, s>>>(q16, k16v, vt, src, dst, attnO);
  // H = LN(q + O) -> q32 (in place) and h16
  add_ln_k<<<LE, 32, 0, s>>>(q32, attnO, F(mi.g1), F(mi.b1), q32, h16);
  // F = relu(H @ Wo + bo) -> attnO (reuse)
  launch_gemm(s, G_FFN, h16, F(mi.oW), wtbuf, F(mi.ob), nullptr, attnO, nullptr, LE, DIMC, DIMC);
  // h_next = LN(H + F) -> h32 and h16
  add_ln_k<<<LE, 32, 0, s>>>(q32, attnO, F(mi.g2), F(mi.b2), h32, h16);
}

extern "C" void kernel_launch(void* const* d_in, const int* in_sizes, int n_in,
                              void* d_out, int out_size, void* d_ws, size_t ws_size,
                              hipStream_t stream)
{
  (void)in_sizes; (void)n_in; (void)out_size; (void)ws_size;

  const float* x     = (const float*)d_in[0];
  const float* ea    = (const float*)d_in[1];
  const int*   eidx  = (const int*)d_in[2];
  const int*   batch = (const int*)d_in[3];
  const int* src = eidx;
  const int* dst = eidx + LE;

  char* p = (char*)d_ws;
  auto carve = [&](size_t bytes) {
    void* r = (void*)p; p += (bytes + 255) & ~size_t(255); return r;
  };
  float*    nodeF  = (float*)carve((size_t)NN * HIDC * 4);
  float*    h32    = (float*)carve((size_t)LE * DIMC * 4);
  _Float16* h16    = (_Float16*)carve((size_t)LE * DIMC * 2);
  float*    q32    = (float*)carve((size_t)LE * DIMC * 4);
  _Float16* q16    = (_Float16*)carve((size_t)LE * DIMC * 2);
  _Float16* k16v   = (_Float16*)carve((size_t)LE * DIMC * 2);
  _Float16* vt     = (_Float16*)carve((size_t)DIMC * LE * 2);   // V transposed
  float*    attnO  = (float*)carve((size_t)LE * DIMC * 4);
  _Float16* wtbuf  = (_Float16*)carve((size_t)DIMC * DIMC * 2); // f16 W^T staging
  int*      ebatch = (int*)carve((size_t)LE * 4);
  float*    qpma   = (float*)carve((size_t)DIMC * 4);
  float*    Opma   = (float*)carve((size_t)NGR * DIMC * 4);

  // Encoders
  node_enc_k<<<NN, HIDC, 0, stream>>>(x, (const float*)d_in[10],
                                      (const float*)d_in[11], nodeF);
  edge_enc_k<<<LE, DIMC, 0, stream>>>(ea, (const float*)d_in[8],
                                      (const float*)d_in[9], nodeF,
                                      src, dst, batch, h32, h16, ebatch);

  // Two SAB layers (param leaf indices per sorted-pytree flatten order)
  MabIdx s1 = {25, 26, 27, 28, 29, 30, 31, 32, 33, 34, 35, 36};
  MabIdx s2 = {37, 38, 39, 40, 41, 42, 43, 44, 45, 46, 47, 48};
  run_sab(stream, d_in, s1, h32, h16, q32, q16, k16v, vt, wtbuf, attnO, src, dst);
  run_sab(stream, d_in, s2, h32, h16, q32, q16, k16v, vt, wtbuf, attnO, src, dst);

  // PMA pooling (pma params: b1=12 b2=13 g1=14 g2=15 kW=16 kb=17 oW=18 ob=19
  //              qW=20 qb=21 seed=22 vW=23 vb=24)
  pma_q_k<<<1, DIMC, 0, stream>>>((const float*)d_in[22], (const float*)d_in[20],
                                  (const float*)d_in[21], qpma);
  launch_gemm(stream, G_K,  h16, (const float*)d_in[16], wtbuf,
              (const float*)d_in[17], k16v, nullptr, nullptr, LE, DIMC, DIMC);
  launch_gemm(stream, G_VT, h16, (const float*)d_in[23], wtbuf,
              (const float*)d_in[24], nullptr, nullptr, vt, LE, DIMC, DIMC);
  pma_attn_k<<<dim3(NGR, NHEAD), 32, 0, stream>>>(qpma, k16v, vt, ebatch, Opma);

  // Tail: LN/FFN/LN + classifier (cls1: W=4 b=5, cls2: W=6 b=7)
  pma_tail_k<<<NGR, 256, 0, stream>>>(
      qpma, Opma,
      (const float*)d_in[14], (const float*)d_in[12],   // g1, b1
      (const float*)d_in[15], (const float*)d_in[13],   // g2, b2
      (const float*)d_in[18], (const float*)d_in[19],   // o.W, o.b
      (const float*)d_in[4],  (const float*)d_in[5],    // cls1.W, cls1.b
      (const float*)d_in[6],  (const float*)d_in[7],    // cls2.W, cls2.b
      (float*)d_out);
}
